// Net1_72696616452505
// MI455X (gfx1250) — compile-verified, matrix operand order checked by reading, not run on hardware
//
#include <hip/hip_runtime.h>
#include <hip/hip_bf16.h>
#include <math.h>

typedef __attribute__((ext_vector_type(16))) __bf16 v16bf;
typedef __attribute__((ext_vector_type(8)))  __bf16 v8bf;
typedef __attribute__((ext_vector_type(8)))  float  v8f;
typedef __attribute__((ext_vector_type(4)))  float  v4f;

// ---------------- workspace layout (bytes) ----------------
// [0, 524288)          : Bpk   bf16[4096*64]  rel_table packed to WMMA B layout
// [524288, 532480)     : Gwpk  bf16[64*64]    graph_weight packed to WMMA B layout
// [532480, 534528)     : pooledSum f32[8*64]  (sum over n, divide by 4096 later)
// [534528, 536576)     : treeEmb   f32[8*64]
#define OFF_GWPK   524288
#define OFF_POOLED 532480
#define OFF_TREE   534528

// WMMA bf16 operand element->K mapping: lane kbase=(lane>>4)*8,
// element j (0..15) -> K = kbase + j + (j>=8 ? 8 : 0)

// ---------------- prep: pack B operands (bf16), zero accumulators ----------
__global__ __launch_bounds__(256) void prep_kernel(
    const float* __restrict__ rel, const float* __restrict__ gw,
    __bf16* __restrict__ Bpk, __bf16* __restrict__ Gwpk,
    float* __restrict__ pooledSum, float* __restrict__ treeEmb)
{
    int idx = blockIdx.x * 256 + threadIdx.x;
    if (idx < 4096 * 64) {   // [kc(128)][nt(4)][lane(32)][j(16)]
        int j    = idx & 15;
        int lane = (idx >> 4) & 31;
        int nt   = (idx >> 9) & 3;
        int kc   = idx >> 11;
        int kbase = (lane >> 4) * 8;
        int k = kc * 32 + kbase + j + (j >= 8 ? 8 : 0);
        int n = nt * 16 + (lane & 15);
        Bpk[idx] = (__bf16)rel[k * 64 + n];
    }
    if (idx < 64 * 64) {     // [kc2(2)][nt(4)][lane(32)][j(16)]
        int j    = idx & 15;
        int lane = (idx >> 4) & 31;
        int nt   = (idx >> 9) & 3;
        int kc   = idx >> 11;   // 0..1
        int kbase = (lane >> 4) * 8;
        int k = kc * 32 + kbase + j + (j >= 8 ? 8 : 0);
        int n = nt * 16 + (lane & 15);
        Gwpk[idx] = (__bf16)gw[k * 64 + n];
    }
    if (idx < 512) { pooledSum[idx] = 0.0f; treeEmb[idx] = 0.0f; }
}

// ---------------- fused: support = link@rel ; pooledSum += relu(support@Gw+b)
// grid: 8 batches * 32 m-tiles ; block: 256 threads (8 waves), wave owns 16 rows
__global__ __launch_bounds__(256) void gemm_pooled_kernel(
    const float* __restrict__ link, const __bf16* __restrict__ Bpk,
    const __bf16* __restrict__ Gwpk, const float* __restrict__ gbias,
    float* __restrict__ pooledSum)
{
    __shared__ __align__(32) __bf16 Bst[2][2048];     // 2 x 4KB double buffer
    __shared__ float Sup[8][16][66];                  // per-wave support tile

    const int b    = blockIdx.x >> 5;
    const int mt   = blockIdx.x & 31;
    const int tid  = threadIdx.x;
    const int wave = tid >> 5;
    const int lane = tid & 31;
    const int lcol = lane & 15;
    const int kbase = (lane >> 4) * 8;

    const int mrow = mt * 128 + wave * 16 + lcol;
    const float* __restrict__ arow =
        link + ((size_t)b << 24) + (size_t)mrow * 4096;

    v8f acc0 = {}, acc1 = {}, acc2v = {}, acc3 = {};

    // stage chunk 0
    {
        const v8bf* src = (const v8bf*)Bpk;
        ((v8bf*)&Bst[0][0])[tid] = src[tid];
    }
    __syncthreads();

    for (int kc = 0; kc < 128; ++kc) {
        const int cur = kc & 1;
        if (kc + 1 < 128) {  // stage next chunk into the other buffer
            const v8bf* src = (const v8bf*)(Bpk + (size_t)(kc + 1) * 2048);
            ((v8bf*)&Bst[cur ^ 1][0])[tid] = src[tid];
        }
        // A fragment: 16 fp32 -> bf16 (row mrow, K = kc*32 + kbase + {0..7,16..23})
        const float* ap = arow + kc * 32 + kbase;
        __builtin_prefetch(ap + 4 * 32, 0, 1);   // global_prefetch_b8, 4 chunks ahead
        v4f x0 = *(const v4f*)(ap);
        v4f x1 = *(const v4f*)(ap + 4);
        v4f x2 = *(const v4f*)(ap + 16);
        v4f x3 = *(const v4f*)(ap + 20);
        v16bf af;
#pragma unroll
        for (int j = 0; j < 4; ++j) {
            af[j]      = (__bf16)x0[j];
            af[4 + j]  = (__bf16)x1[j];
            af[8 + j]  = (__bf16)x2[j];
            af[12 + j] = (__bf16)x3[j];
        }
        v16bf b0 = *(const v16bf*)&Bst[cur][(0 * 32 + lane) * 16];
        v16bf b1 = *(const v16bf*)&Bst[cur][(1 * 32 + lane) * 16];
        v16bf b2 = *(const v16bf*)&Bst[cur][(2 * 32 + lane) * 16];
        v16bf b3 = *(const v16bf*)&Bst[cur][(3 * 32 + lane) * 16];
        acc0  = __builtin_amdgcn_wmma_f32_16x16x32_bf16(false, af, false, b0, (short)0, acc0,  false, false);
        acc1  = __builtin_amdgcn_wmma_f32_16x16x32_bf16(false, af, false, b1, (short)0, acc1,  false, false);
        acc2v = __builtin_amdgcn_wmma_f32_16x16x32_bf16(false, af, false, b2, (short)0, acc2v, false, false);
        acc3  = __builtin_amdgcn_wmma_f32_16x16x32_bf16(false, af, false, b3, (short)0, acc3,  false, false);
        __syncthreads();
    }

    // ---- epilogue: spill support tile to LDS (C layout -> row/col) ----
    const int lhalf = lane >> 4;
#pragma unroll
    for (int v = 0; v < 8; ++v) {
        Sup[wave][v + 8 * lhalf][0 * 16 + lcol] = acc0[v];
        Sup[wave][v + 8 * lhalf][1 * 16 + lcol] = acc1[v];
        Sup[wave][v + 8 * lhalf][2 * 16 + lcol] = acc2v[v];
        Sup[wave][v + 8 * lhalf][3 * 16 + lcol] = acc3[v];
    }
    __syncthreads();

    // ---- second matmul: relu(support(16x64) @ Gw(64x64) + bias) ----
    v8f o0 = {}, o1 = {}, o2 = {}, o3 = {};
#pragma unroll
    for (int kc2 = 0; kc2 < 2; ++kc2) {
        v16bf af2;
#pragma unroll
        for (int j = 0; j < 16; ++j) {
            int e = kc2 * 32 + kbase + j + (j >= 8 ? 8 : 0);
            af2[j] = (__bf16)Sup[wave][lcol][e];
        }
        v16bf g0 = *(const v16bf*)(Gwpk + (size_t)((kc2 * 4 + 0) * 32 + lane) * 16);
        v16bf g1 = *(const v16bf*)(Gwpk + (size_t)((kc2 * 4 + 1) * 32 + lane) * 16);
        v16bf g2 = *(const v16bf*)(Gwpk + (size_t)((kc2 * 4 + 2) * 32 + lane) * 16);
        v16bf g3 = *(const v16bf*)(Gwpk + (size_t)((kc2 * 4 + 3) * 32 + lane) * 16);
        o0 = __builtin_amdgcn_wmma_f32_16x16x32_bf16(false, af2, false, g0, (short)0, o0, false, false);
        o1 = __builtin_amdgcn_wmma_f32_16x16x32_bf16(false, af2, false, g1, (short)0, o1, false, false);
        o2 = __builtin_amdgcn_wmma_f32_16x16x32_bf16(false, af2, false, g2, (short)0, o2, false, false);
        o3 = __builtin_amdgcn_wmma_f32_16x16x32_bf16(false, af2, false, g3, (short)0, o3, false, false);
    }

    // bias + relu + per-column partial sums -> global atomic accumulate
    float* pb = pooledSum + b * 64;
#pragma unroll
    for (int t = 0; t < 4; ++t) {
        v8f o = (t == 0) ? o0 : (t == 1) ? o1 : (t == 2) ? o2 : o3;
        float bias = gbias[t * 16 + lcol];
        float csum = 0.0f;
#pragma unroll
        for (int v = 0; v < 8; ++v) csum += fmaxf(o[v] + bias, 0.0f);
        atomicAdd(pb + t * 16 + lcol, csum);   // lanes l and l+16 cover the 16 rows
    }
}

// ---------------- tiny attention scan: one WG per (b,t), 64 threads --------
__global__ __launch_bounds__(64) void attn_kernel(
    const float* __restrict__ rel, const float* __restrict__ col,
    const float* __restrict__ Wk, const float* __restrict__ Wq,
    const float* __restrict__ Wv, const int* __restrict__ tids,
    const int* __restrict__ lids, const int* __restrict__ rids,
    float* __restrict__ treeEmb)
{
    const int bt = blockIdx.x;          // 0..63
    const int b  = bt >> 3;
    const int e  = threadIdx.x;         // 0..63

    __shared__ float q[4][64];
    __shared__ float Qs[4][64];
    __shared__ float Ks[4][64];
    __shared__ float sc[4][4];
    __shared__ float at[4][4];

    const int* tid_bt = tids + bt * 17;
    float prev = rel[(size_t)tid_bt[0] * 64 + e];

    for (int d = 0; d < 16; ++d) {
        __syncthreads();
        float rte = rel[(size_t)tid_bt[1 + d] * 64 + e];
        const int* lp = lids + (bt * 16 + d) * 2;
        const int* rp = rids + (bt * 16 + d) * 2;
        float lce = 0.5f * (col[(size_t)lp[0] * 64 + e] + col[(size_t)lp[1] * 64 + e]);
        float rce = 0.5f * (col[(size_t)rp[0] * 64 + e] + col[(size_t)rp[1] * 64 + e]);
        q[0][e] = prev; q[1][e] = lce; q[2][e] = rce; q[3][e] = rte;
        __syncthreads();

        float Kr[4] = {0,0,0,0}, Qr[4] = {0,0,0,0}, Vr[4] = {0,0,0,0};
        for (int f = 0; f < 64; ++f) {
            float wk = Wk[f * 64 + e], wq = Wq[f * 64 + e], wv = Wv[f * 64 + e];
#pragma unroll
            for (int i = 0; i < 4; ++i) {
                float qv = q[i][f];
                Kr[i] = fmaf(qv, wk, Kr[i]);
                Qr[i] = fmaf(qv, wq, Qr[i]);
                Vr[i] = fmaf(qv, wv, Vr[i]);
            }
        }
#pragma unroll
        for (int i = 0; i < 4; ++i) { Qs[i][e] = Qr[i]; Ks[i][e] = Kr[i]; }
        __syncthreads();

        if (e < 16) {
            int i = e >> 2, jj = e & 3;
            float s = 0.0f;
            for (int f = 0; f < 64; ++f) s = fmaf(Qs[i][f], Ks[jj][f], s);
            sc[i][jj] = s * (1.0f / 64.0f);
        }
        __syncthreads();

        if (e < 4) {
            float m = sc[e][0];
#pragma unroll
            for (int j = 1; j < 4; ++j) m = fmaxf(m, sc[e][j]);
            float ex[4], sum = 0.0f;
#pragma unroll
            for (int j = 0; j < 4; ++j) { ex[j] = expf(sc[e][j] - m); sum += ex[j]; }
            float inv = 1.0f / sum;
#pragma unroll
            for (int j = 0; j < 4; ++j) at[e][j] = ex[j] * inv;
        }
        __syncthreads();

        float out = 0.0f;
#pragma unroll
        for (int j = 0; j < 4; ++j) {
            float cs = at[0][j] + at[1][j] + at[2][j] + at[3][j];
            out = fmaf(cs, Vr[j], out);
        }
        prev = out;
    }
    atomicAdd(treeEmb + b * 64 + e, prev);
}

// ---------------- logits = clip(log(mask)) + [pooled||tree] @ fc_w ----------
__global__ __launch_bounds__(256) void logits_kernel(
    const float* __restrict__ amask, const float* __restrict__ fc,
    const float* __restrict__ pooledSum, const float* __restrict__ treeEmb,
    float* __restrict__ out)
{
    int idx = blockIdx.x * 256 + threadIdx.x;   // 0..32767
    int b = idx >> 12, a = idx & 4095;
    float acc = 0.0f;
#pragma unroll 4
    for (int k = 0; k < 64; ++k)
        acc = fmaf(pooledSum[b * 64 + k] * (1.0f / 4096.0f), fc[(size_t)k * 4096 + a], acc);
#pragma unroll 4
    for (int k = 0; k < 64; ++k)
        acc = fmaf(treeEmb[b * 64 + k], fc[(size_t)(64 + k) * 4096 + a], acc);
    float lm = logf(amask[idx]);
    lm = fminf(fmaxf(lm, -3.4e38f), 3.4e38f);
    out[idx] = lm + acc;
}

extern "C" void kernel_launch(void* const* d_in, const int* in_sizes, int n_in,
                              void* d_out, int out_size, void* d_ws, size_t ws_size,
                              hipStream_t stream) {
    const float* link  = (const float*)d_in[0];
    const float* amask = (const float*)d_in[1];
    const float* col   = (const float*)d_in[2];
    const float* rel   = (const float*)d_in[3];
    const float* gw    = (const float*)d_in[4];
    const float* gb    = (const float*)d_in[5];
    const float* Wk    = (const float*)d_in[6];
    const float* Wq    = (const float*)d_in[7];
    const float* Wv    = (const float*)d_in[8];
    const float* fc    = (const float*)d_in[9];
    const int* tids    = (const int*)d_in[10];
    const int* lids    = (const int*)d_in[11];
    const int* rids    = (const int*)d_in[12];

    char* ws = (char*)d_ws;
    __bf16* Bpk       = (__bf16*)ws;
    __bf16* Gwpk      = (__bf16*)(ws + OFF_GWPK);
    float*  pooledSum = (float*)(ws + OFF_POOLED);
    float*  treeEmb   = (float*)(ws + OFF_TREE);
    float*  out       = (float*)d_out;

    prep_kernel<<<1024, 256, 0, stream>>>(rel, gw, Bpk, Gwpk, pooledSum, treeEmb);
    gemm_pooled_kernel<<<256, 256, 0, stream>>>(link, Bpk, Gwpk, gb, pooledSum);
    attn_kernel<<<64, 64, 0, stream>>>(rel, col, Wk, Wq, Wv, tids, lids, rids, treeEmb);
    logits_kernel<<<128, 256, 0, stream>>>(amask, fc, pooledSum, treeEmb, out);
}
